// FixedRateVectorQuantizer_56848187129908
// MI455X (gfx1250) — compile-verified, matrix operand order checked by reading, not run on hardware
//
#include <hip/hip_runtime.h>
#include <math.h>

typedef __attribute__((ext_vector_type(16))) _Float16 v16h;
typedef __attribute__((ext_vector_type(8)))  float    v8f;
typedef unsigned int u32x4 __attribute__((ext_vector_type(4)));
typedef int          i32x8 __attribute__((ext_vector_type(8)));
typedef int          i32x4 __attribute__((ext_vector_type(4)));

#define NROW 131072
#define DDIM 128
#define PCB  512
#define BPITCH  136   // padded LDS row pitch in halfs (128 + 8)
#define BPITCHW 68    // same in dwords
#define COMMIT_W 0.1f
#define ENT_LAMBDA 200000.0f

// ---------------------------------------------------------------- mu ----
__global__ __launch_bounds__(128) void vq_mu_kernel(const float* __restrict__ cb,
                                                    float* __restrict__ mu) {
  int d = threadIdx.x;
  float s = 0.0f;
  for (int p = 0; p < PCB; ++p) s += cb[p * DDIM + d];
  mu[d] = s * (1.0f / (float)PCB);
}

// --------------------------------------------------------------- cov ----
__global__ __launch_bounds__(256) void vq_cov_kernel(const float* __restrict__ cb,
                                                     const float* __restrict__ mu,
                                                     float* __restrict__ cov) {
  int t = blockIdx.x * 256 + threadIdx.x;   // 0 .. 16383
  int i = t >> 7, j = t & 127;
  float mi = mu[i], mj = mu[j];
  float s = 0.0f;
  for (int p = 0; p < PCB; ++p)
    s += (cb[p * DDIM + i] - mi) * (cb[p * DDIM + j] - mj);
  float v = s * (1.0f / (float)(PCB - 1));
  if (i == j) v += 1e-3f;
  cov[t] = v;
}

// --------------------------------- in-place Gauss-Jordan inverse --------
__global__ __launch_bounds__(256) void vq_invert_kernel(const float* __restrict__ covIn,
                                                        float* __restrict__ invOut,
                                                        _Float16* __restrict__ invOutH) {
  __shared__ float A[DDIM * DDIM];          // 64KB
  int tid = threadIdx.x;
  for (int i = tid; i < DDIM * DDIM; i += 256) A[i] = covIn[i];
  __syncthreads();

  float f[64];
  for (int k = 0; k < DDIM; ++k) {
    float piv = 1.0f / A[k * DDIM + k];
#pragma unroll 8
    for (int s = 0; s < 64; ++s) {
      int c = tid + 256 * s;
      f[s] = A[(c >> 7) * DDIM + k];
    }
    __syncthreads();
    if (tid < DDIM)
      A[k * DDIM + tid] = (tid == k) ? piv : A[k * DDIM + tid] * piv;
    __syncthreads();
#pragma unroll 8
    for (int s = 0; s < 64; ++s) {
      int c = tid + 256 * s;
      int i = c >> 7, j = c & 127;
      if (i != k) {
        float base = (j == k) ? 0.0f : A[c];
        A[c] = base - f[s] * A[k * DDIM + j];
      }
    }
    __syncthreads();
  }
  for (int i = tid; i < DDIM * DDIM; i += 256) {
    float v = A[i];
    invOut[i] = v;
    invOutH[i] = (_Float16)v;
  }
}

// ------------------------------- cS = C @ invCov (f16), cq = diag ------
__global__ __launch_bounds__(128) void vq_prep_kernel(const float* __restrict__ cb,
                                                      const float* __restrict__ S,
                                                      _Float16* __restrict__ cSh,
                                                      float* __restrict__ cq) {
  __shared__ float crow[DDIM];
  __shared__ float part[DDIM];
  int n = blockIdx.x, k = threadIdx.x;
  crow[k] = cb[n * DDIM + k];
  __syncthreads();
  float s = 0.0f;
  const float* Sr = S + k * DDIM;           // symmetric
  for (int j = 0; j < DDIM; ++j) s += crow[j] * Sr[j];
  cSh[n * DDIM + k] = (_Float16)s;
  part[k] = s * crow[k];
  __syncthreads();
  for (int off = 64; off > 0; off >>= 1) {
    if (k < off) part[k] += part[k + off];
    __syncthreads();
  }
  if (k == 0) cq[n] = part[0];
}

// ---------------------------------------------------------- TDM util ----
__device__ static inline unsigned ldsOffset(const void* p) {
  return (unsigned)(unsigned long long)(size_t)p;
}

#if __has_builtin(__builtin_amdgcn_tensor_load_to_lds)
// 2D f16 tile DMA: rows x cols (halfs), row-major, TDM inserts 4 dwords of
// padding after every 64 dwords (=1 row) -> LDS pitch 136 halfs.
__device__ static inline void tdm_load_2d_f16(const void* gptr, unsigned ldsOff,
                                              int rows, int cols) {
  unsigned long long ga = (unsigned long long)(size_t)gptr;
  u32x4 g0;
  g0[0] = 1u;                                           // count=1, user mode
  g0[1] = ldsOff;                                       // lds_addr
  g0[2] = (unsigned)(ga & 0xffffffffull);               // global_addr lo
  g0[3] = (unsigned)((ga >> 32) & 0x1ffffffull) | (2u << 30);  // hi + type=2
  i32x8 g1;
  g1[0] = (1 << 16)        // data_size = 2 bytes
        | (1 << 20)        // pad_enable
        | (5 << 22)        // pad_interval: 64 dwords
        | (3 << 25);       // pad_amount: 4 dwords
  g1[1] = (cols & 0xffff) << 16;                        // tensor_dim0 lo16
  g1[2] = ((rows & 0xffff) << 16) | ((cols >> 16) & 0xffff); // dim1 lo | dim0 hi
  g1[3] = ((cols & 0xffff) << 16) | ((rows >> 16) & 0xffff); // tile0  | dim1 hi
  g1[4] = (rows & 0xffff);                              // tile_dim1 (tile_dim2=0)
  g1[5] = cols;                                         // tensor_dim0_stride lo32
  g1[6] = 0;
  g1[7] = 0;
  i32x4 z4 = {0, 0, 0, 0};
  i32x8 z8 = {0, 0, 0, 0, 0, 0, 0, 0};
  // amdgpu-toolchain (clang-23) 6-arg form: (g0, g1, g2, g3, g4, cpol)
  __builtin_amdgcn_tensor_load_to_lds(g0, g1, z4, z4, z8, 0);
}
#endif

// ------------------------------------------------ fused main kernel ----
// Block = 8 waves; each wave owns two 16-row M-tiles (32 rows, 256/block).
// cS (512x128 f16) and invCov (128x128 f16) staged in LDS once per block.
__global__ __launch_bounds__(256) void vq_main_kernel(
    const float* __restrict__ X, const float* __restrict__ CB,
    const _Float16* __restrict__ cSg, const _Float16* __restrict__ Sg,
    const float* __restrict__ cq, float* __restrict__ outQ,
    float* __restrict__ outIdx, unsigned* __restrict__ gcount,
    double* __restrict__ gloss) {
  __shared__ _Float16 bsh[PCB * BPITCH];        // 136KB: cS, padded pitch
  __shared__ _Float16 ssh[DDIM * BPITCH];       // 34KB : invCov
  __shared__ _Float16 xsh[8][32 * BPITCH];      // 68KB : per-wave X tiles
  __shared__ float    cqs[PCB];
  __shared__ unsigned hist[PCB];
  __shared__ int      idxs[8][32];
  __shared__ float    lossp[8];

  const int tid = threadIdx.x;
  const int w = tid >> 5, l = tid & 31;
  const int ln = l & 15, h = l >> 4;

  for (int i = tid; i < PCB; i += 256) { hist[i] = 0u; cqs[i] = cq[i]; }

#if __has_builtin(__builtin_amdgcn_tensor_load_to_lds)
  if (w == 0) {
    tdm_load_2d_f16(cSg, ldsOffset(bsh), PCB, DDIM);
    __builtin_amdgcn_s_wait_tensorcnt(0);
  } else if (w == 1) {
    tdm_load_2d_f16(Sg, ldsOffset(ssh), DDIM, DDIM);
    __builtin_amdgcn_s_wait_tensorcnt(0);
  }
#else
  {
    const unsigned* s0 = (const unsigned*)cSg;
    unsigned* d0 = (unsigned*)bsh;
    for (int i = tid; i < PCB * 64; i += 256)
      d0[(i >> 6) * BPITCHW + (i & 63)] = s0[i];
    const unsigned* s1 = (const unsigned*)Sg;
    unsigned* d1 = (unsigned*)ssh;
    for (int i = tid; i < DDIM * 64; i += 256)
      d1[(i >> 6) * BPITCHW + (i & 63)] = s1[i];
  }
#endif

  // stage this wave's 32 X rows into LDS as f16 (padded pitch)
  const long row0 = ((long)blockIdx.x * 8 + w) * 32;
  {
    const float* xg = X + row0 * DDIM;
    for (int r = 0; r < 32; ++r) {
      float4 v = ((const float4*)(xg + r * DDIM))[l];
      _Float16* d = &xsh[w][r * BPITCH + 4 * l];
      d[0] = (_Float16)v.x; d[1] = (_Float16)v.y;
      d[2] = (_Float16)v.z; d[3] = (_Float16)v.w;
    }
  }
  __syncthreads();

  // A fragments: two tiles x four K-32 chunks
  v16h a[2][4];
#pragma unroll
  for (int tt = 0; tt < 2; ++tt)
#pragma unroll
    for (int kc = 0; kc < 4; ++kc) {
      const _Float16* p = &xsh[w][(tt * 16 + ln) * BPITCH + kc * 32 + 8 * h];
#pragma unroll
      for (int t = 0; t < 8; ++t) {
        a[tt][kc][t] = p[t];
        a[tt][kc][8 + t] = p[16 + t];
      }
    }

  // ---- distance GEMM + running argmin over 32 codebook N-tiles.
  // Candidates arrive in ascending n within a lane, so strict '<' keeps
  // first-minimum semantics; selects are branchless (cndmask, no exec churn).
  float bestv[2][8]; int bestn[2][8];
#pragma unroll
  for (int tt = 0; tt < 2; ++tt)
#pragma unroll
    for (int r = 0; r < 8; ++r) { bestv[tt][r] = 3.4e38f; bestn[tt][r] = PCB; }

  for (int nt = 0; nt < 32; ++nt) {
    const int n = nt * 16 + ln;
    v8f acc0 = {}, acc1 = {};
    const _Float16* bp = &bsh[n * BPITCH];
#pragma unroll
    for (int kc = 0; kc < 4; ++kc) {
      v16h b;
      const _Float16* q = bp + kc * 32 + 8 * h;
#pragma unroll
      for (int t = 0; t < 8; ++t) { b[t] = q[t]; b[8 + t] = q[16 + t]; }
      acc0 = __builtin_amdgcn_wmma_f32_16x16x32_f16(false, a[0][kc], false, b,
                                                    (short)0, acc0, false, false);
      acc1 = __builtin_amdgcn_wmma_f32_16x16x32_f16(false, a[1][kc], false, b,
                                                    (short)0, acc1, false, false);
    }
    const float cqn = cqs[n];
#pragma unroll
    for (int r = 0; r < 8; ++r) {
      float v0 = __builtin_fmaf(-2.0f, acc0[r], cqn);
      int c0 = v0 < bestv[0][r];
      bestv[0][r] = c0 ? v0 : bestv[0][r];
      bestn[0][r] = c0 ? n : bestn[0][r];
      float v1 = __builtin_fmaf(-2.0f, acc1[r], cqn);
      int c1 = v1 < bestv[1][r];
      bestv[1][r] = c1 ? v1 : bestv[1][r];
      bestn[1][r] = c1 ? n : bestn[1][r];
    }
  }

  // ---- xq = x S x^T via WMMA against invCov(f16)
  float xq[2][8];
#pragma unroll
  for (int tt = 0; tt < 2; ++tt)
#pragma unroll
    for (int r = 0; r < 8; ++r) xq[tt][r] = 0.0f;

  for (int nt = 0; nt < 8; ++nt) {
    const int n = nt * 16 + ln;
    v8f acc0 = {}, acc1 = {};
    const _Float16* bp = &ssh[n * BPITCH];
#pragma unroll
    for (int kc = 0; kc < 4; ++kc) {
      v16h b;
      const _Float16* q = bp + kc * 32 + 8 * h;
#pragma unroll
      for (int t = 0; t < 8; ++t) { b[t] = q[t]; b[8 + t] = q[16 + t]; }
      acc0 = __builtin_amdgcn_wmma_f32_16x16x32_f16(false, a[0][kc], false, b,
                                                    (short)0, acc0, false, false);
      acc1 = __builtin_amdgcn_wmma_f32_16x16x32_f16(false, a[1][kc], false, b,
                                                    (short)0, acc1, false, false);
    }
#pragma unroll
    for (int r = 0; r < 8; ++r) {
      int mm = r + 8 * h;
      xq[0][r] += acc0[r] * (float)xsh[w][(mm)      * BPITCH + n];
      xq[1][r] += acc1[r] * (float)xsh[w][(16 + mm) * BPITCH + n];
    }
  }

  // ---- reduce across the 16-lane N-group (xor bits 0..3 only);
  // keep index tie-break here, still branchless.
#pragma unroll
  for (int off = 1; off < 16; off <<= 1) {
#pragma unroll
    for (int tt = 0; tt < 2; ++tt)
#pragma unroll
      for (int r = 0; r < 8; ++r) {
        float ov = __shfl_xor(bestv[tt][r], off, 32);
        int   oi = __shfl_xor(bestn[tt][r], off, 32);
        int c = (ov < bestv[tt][r]) |
                ((ov == bestv[tt][r]) & (oi < bestn[tt][r]));
        bestv[tt][r] = c ? ov : bestv[tt][r];
        bestn[tt][r] = c ? oi : bestn[tt][r];
        xq[tt][r] += __shfl_xor(xq[tt][r], off, 32);
      }
  }

  float myloss = 0.0f;
  if (ln == 0) {
#pragma unroll
    for (int tt = 0; tt < 2; ++tt)
#pragma unroll
      for (int r = 0; r < 8; ++r) {
        int mm = tt * 16 + r + 8 * h;
        long row = row0 + mm;
        int n = bestn[tt][r];
        outIdx[row] = (float)n;
        idxs[w][mm] = n;
        atomicAdd(&hist[n], 1u);
        myloss += bestv[tt][r] + xq[tt][r];   // (q-x) S (q-x)^T
      }
  }
  myloss += __shfl_xor(myloss, 16, 32);
  if (l == 0) lossp[w] = myloss;
  __syncthreads();

  // ---- gather quantized rows (full f32 fidelity)
  for (int mm = 0; mm < 32; ++mm) {
    int n = idxs[w][mm];
    long row = row0 + mm;
    float4 qv = ((const float4*)(CB + (long)n * DDIM))[l];
    ((float4*)(outQ + row * DDIM))[l] = qv;
  }

  if (tid == 0) {
    float s = 0.0f;
    for (int i = 0; i < 8; ++i) s += lossp[i];
    atomicAdd(gloss, (double)s);
  }
  for (int i = tid; i < PCB; i += 256)
    if (hist[i]) atomicAdd(&gcount[i], hist[i]);
}

// ----------------------------------------------------------- finalize --
__global__ void vq_finalize_kernel(const unsigned* __restrict__ cnt,
                                   const double* __restrict__ loss,
                                   float* __restrict__ outLoss) {
  if (threadIdx.x == 0 && blockIdx.x == 0) {
    float tot = 0.0f;
    for (int i = 0; i < PCB; ++i) tot += (float)cnt[i];
    float H = 0.0f;
    for (int i = 0; i < PCB; ++i) {
      float p = (float)cnt[i] / (tot + 1e-8f);
      H -= p * logf(p + 1e-8f);
    }
    float mean = (float)(*loss) / (float)NROW;
    outLoss[0] = COMMIT_W * mean;                 // commitment_loss
    outLoss[1] = mean - ENT_LAMBDA * H;           // codebook_loss
  }
}

// -------------------------------------------------------------- launch --
extern "C" void kernel_launch(void* const* d_in, const int* in_sizes, int n_in,
                              void* d_out, int out_size, void* d_ws, size_t ws_size,
                              hipStream_t stream) {
  const float* X  = (const float*)d_in[0];
  const float* CB = (const float*)d_in[1];
  float* out = (float*)d_out;
  float* ws  = (float*)d_ws;

  float*    cov  = ws;                                  // 16384
  float*    invS = ws + 16384;                          // 16384
  _Float16* invH = (_Float16*)(ws + 32768);             // 16384 halfs
  _Float16* cSh  = (_Float16*)(ws + 40960);             // 65536 halfs
  float*    cq   = ws + 73728;                          // 512
  float*    mu   = ws + 74240;                          // 128
  double*   gloss = (double*)(ws + 74368);
  unsigned* gcnt  = (unsigned*)(ws + 74372);

  (void)hipMemsetAsync(ws + 74368, 0, 4096, stream);

  vq_mu_kernel    <<<1,   128, 0, stream>>>(CB, mu);
  vq_cov_kernel   <<<64,  256, 0, stream>>>(CB, mu, cov);
  vq_invert_kernel<<<1,   256, 0, stream>>>(cov, invS, invH);
  vq_prep_kernel  <<<PCB, 128, 0, stream>>>(CB, invS, cSh, cq);

  float* outQ    = out;
  float* outLoss = out + (long)NROW * DDIM;
  float* outIdx  = outLoss + 2;

  vq_main_kernel<<<NROW / 256, 256, 0, stream>>>(X, CB, cSh, invH, cq,
                                                 outQ, outIdx, gcnt, gloss);
  vq_finalize_kernel<<<1, 32, 0, stream>>>(gcnt, gloss, outLoss);
}
